// TensorizedEmbedding_order4_369367188185
// MI455X (gfx1250) — matrix operand order repaired and checked by reading, not compiled, not run on hardware
//
#include <hip/hip_runtime.h>

typedef __attribute__((ext_vector_type(16))) _Float16 v16h;
typedef __attribute__((ext_vector_type(8)))  _Float16 v8h;
typedef __attribute__((ext_vector_type(8)))  float    v8f;

// Tables in workspace:
//   w1: out1 as f16, layout [row(200)][m(24)][q(64)]          (row-major)
//   w2: out2 as f16, layout [col(160)][n(32)][q(64)]          (transposed: n-major, q contiguous)
#define W1_ELEMS (200 * 24 * 64)
#define W2_ELEMS (160 * 32 * 64)

// out1[p=n*25+m][j=a*6+b][r] = sum_q f0[0,n,a,q] * f1[q,m,b,r]
__global__ void build_out1(const float* __restrict__ f0, const float* __restrict__ f1,
                           _Float16* __restrict__ w1) {
    int idx = blockIdx.x * blockDim.x + threadIdx.x;   // 307200 threads
    int r = idx & 63;
    int j = (idx >> 6) % 24;
    int p = idx / (64 * 24);
    int a = j / 6, b = j % 6;
    int n = p / 25, m = p % 25;
    const float* f0p = f0 + (n * 4 + a) * 64;                  // f0[0][n][a][q]
    const float* f1p = f1 + (m * 6 + b) * 64 + r;              // f1[q][m][b][r], q-stride 9600
    float acc = 0.f;
#pragma unroll
    for (int q = 0; q < 64; ++q)
        acc += f0p[q] * f1p[q * (25 * 6 * 64)];
    w1[idx] = (_Float16)acc;
}

// out2T[col=n2*8+m2][d=a2*8+b2][q] = sum_s f2[q,n2,a2,s] * f3[s,m2,b2,0]
__global__ void build_out2(const float* __restrict__ f2, const float* __restrict__ f3,
                           _Float16* __restrict__ w2) {
    int idx = blockIdx.x * blockDim.x + threadIdx.x;   // 327680 threads: [col][d][q]
    int q   = idx & 63;
    int d   = (idx >> 6) & 31;
    int col = idx >> 11;
    int a2 = d >> 3, b2 = d & 7;
    int n2 = col >> 3, m2 = col & 7;
    const float* f2p = f2 + ((q * 20 + n2) * 4 + a2) * 64;     // f2[q][n2][a2][s]
    const float* f3p = f3 + (m2 * 8 + b2);                     // f3[s][m2][b2], s-stride 64
    float acc = 0.f;
#pragma unroll
    for (int s = 0; s < 64; ++s)
        acc += f2p[s] * f3p[s * 64];
    w2[idx] = (_Float16)acc;
}

// One wave32 per token: Y(24x32) = A(24x64) * B(64x32), 2x2x2 tiles of
// v_wmma_f32_16x16x32_f16. Grid is exact (65536 waves), EXEC all-ones at WMMA.
__global__ void __launch_bounds__(256)
embed_wmma(const int* __restrict__ x,
           const _Float16* __restrict__ w1,
           const _Float16* __restrict__ w2,
           float* __restrict__ out) {
    const int lane  = threadIdx.x & 31;
    const int wave  = threadIdx.x >> 5;
    const int t     = blockIdx.x * 8 + wave;

    int id = x[t];
    id = (id < 0) ? 0 : ((id >= 32000) ? 31999 : id);
    const int row = id / 160;
    const int col = id % 160;

    const _Float16* Abase = w1 + row * (24 * 64);
    const _Float16* Bbase = w2 + col * (32 * 64);

    const int mHalf = lane >> 4;   // 0: lanes 0-15, 1: lanes 16-31
    const int lmod  = lane & 15;

    v8f acc[2][2] = {};

#pragma unroll
    for (int ks = 0; ks < 64; ks += 32) {
        // A fragment (16x32 f16): lane holds row m, K chunks [kb..kb+7] and [kb+16..kb+23], kb = mHalf*8
        v16h afrag[2];
#pragma unroll
        for (int mt = 0; mt < 2; ++mt) {
            int m = mt * 16 + lmod;
            if (m > 23) m = 23;                       // clamp edge rows (results never stored)
            const v8h* p = (const v8h*)(Abase + m * 64 + ks + mHalf * 8);
            v8h c0 = p[0];                            // K = kb .. kb+7
            v8h c1 = p[2];                            // K = kb+16 .. kb+23
            afrag[mt] = __builtin_shufflevector(c0, c1, 0,1,2,3,4,5,6,7,8,9,10,11,12,13,14,15);
        }
        // B fragment (32x16 f16): lane holds column n, K chunks [kb..kb+7] and [kb+16..kb+23], kb = mHalf*8
        v16h bfrag[2];
#pragma unroll
        for (int nt = 0; nt < 2; ++nt) {
            const v8h* p = (const v8h*)(Bbase + (nt * 16 + lmod) * 64 + ks + mHalf * 8);
            v8h c0 = p[0];
            v8h c1 = p[2];
            bfrag[nt] = __builtin_shufflevector(c0, c1, 0,1,2,3,4,5,6,7,8,9,10,11,12,13,14,15);
        }
#pragma unroll
        for (int mt = 0; mt < 2; ++mt)
#pragma unroll
            for (int nt = 0; nt < 2; ++nt)
                acc[mt][nt] = __builtin_amdgcn_wmma_f32_16x16x32_f16(
                    false, afrag[mt], false, bfrag[nt],
                    (short)0, acc[mt][nt], false, false);
    }

    // D layout: VGPR j -> M = mt*16 + mHalf*8 + j, N = nt*16 + lmod.
    float* obase = out + (size_t)t * 768;

    // mt = 0 tile: all 16 M rows valid (m = mHalf*8 + j in 0..15) -> unconditional stores
#pragma unroll
    for (int nt = 0; nt < 2; ++nt) {
#pragma unroll
        for (int j = 0; j < 8; ++j) {
            int m = mHalf * 8 + j;
            __builtin_nontemporal_store(acc[0][nt][j], obase + m * 32 + nt * 16 + lmod);
        }
    }

    // mt = 1 tile: rows 16 + mHalf*8 + j; valid (< 24) iff mHalf == 0.
    // Uniform per half-wave -> single exec-mask region with 16 clustered stores.
    if (mHalf == 0) {
#pragma unroll
        for (int nt = 0; nt < 2; ++nt) {
#pragma unroll
            for (int j = 0; j < 8; ++j) {
                __builtin_nontemporal_store(acc[1][nt][j],
                                            obase + (16 + j) * 32 + nt * 16 + lmod);
            }
        }
    }
}

extern "C" void kernel_launch(void* const* d_in, const int* in_sizes, int n_in,
                              void* d_out, int out_size, void* d_ws, size_t ws_size,
                              hipStream_t stream) {
    (void)in_sizes; (void)n_in; (void)out_size; (void)ws_size;
    const int*   x  = (const int*)d_in[0];
    const float* f0 = (const float*)d_in[1];
    const float* f1 = (const float*)d_in[2];
    const float* f2 = (const float*)d_in[3];
    const float* f3 = (const float*)d_in[4];
    float* out = (float*)d_out;

    _Float16* w1 = (_Float16*)d_ws;
    _Float16* w2 = (_Float16*)((char*)d_ws + (size_t)W1_ELEMS * sizeof(_Float16));

    build_out1<<<W1_ELEMS / 256, 256, 0, stream>>>(f0, f1, w1);
    build_out2<<<W2_ELEMS / 256, 256, 0, stream>>>(f2, f3, w2);

    // 16*4096 = 65536 tokens, 8 waves/block -> 8192 blocks (exact, no tail)
    embed_wmma<<<8192, 256, 0, stream>>>(x, w1, w2, out);
}
